// LSTM_88983132439001
// MI455X (gfx1250) — matrix operand order repaired and checked by reading, not compiled, not run on hardware
//
#include <hip/hip_runtime.h>

#define B_  256
#define T_  1024
#define D_  64
#define H_  128
#define NG  512          // 4*H
#define XS_STRIDE 72     // 64 + 8 bf16 pad  (144B rows, 16B aligned)
#define HS_STRIDE 136    // 128 + 8 bf16 pad (272B rows, 16B aligned)
#define GS  516          // gates row stride in floats

typedef __bf16 bf16_t;
typedef __attribute__((ext_vector_type(16))) __bf16 v16bf;
typedef __attribute__((ext_vector_type(8)))  __bf16 v8bf;
typedef __attribute__((ext_vector_type(8)))  float  v8f;
typedef __attribute__((ext_vector_type(2)))  float  v2f;   // clang vector: OK for nontemporal builtins

#define L2E 1.442695040888963f

static __device__ __forceinline__ bf16_t f2bf(float f) {
    unsigned int u = __builtin_bit_cast(unsigned int, f);
    unsigned int r = u + 0x7FFFu + ((u >> 16) & 1u);   // round-to-nearest-even
    unsigned short h = (unsigned short)(r >> 16);
    return __builtin_bit_cast(bf16_t, h);
}

// Branch-free activations on hardware transcendentals (v_exp_f32 / v_rcp_f32).
// Saturation falls out of IEEE inf/0 handling: no range checks, no EXEC games.
static __device__ __forceinline__ float fast_sigmoid(float x) {
    return __builtin_amdgcn_rcpf(1.f + __builtin_amdgcn_exp2f(-L2E * x));
}
static __device__ __forceinline__ float fast_tanh(float x) {
    float e = __builtin_amdgcn_exp2f((2.f * L2E) * x);   // e^(2x)
    return 1.f - 2.f * __builtin_amdgcn_rcpf(e + 1.f);
}

// ---------------------------------------------------------------------------
// Prep: transpose + downconvert weights to bf16 in d_ws.
//   WiT : [NG][D]  (n-major)   WhT : [NG][H]  (n-major)
// ---------------------------------------------------------------------------
__global__ void lstm_prep(const float* __restrict__ Wi, const float* __restrict__ Wh,
                          bf16_t* __restrict__ WiT, bf16_t* __restrict__ WhT) {
    int i = blockIdx.x * blockDim.x + threadIdx.x;
    if (i < D_ * NG) {                       // Wi is [D][NG] row-major
        int k = i / NG, n = i % NG;
        WiT[n * D_ + k] = f2bf(Wi[i]);
    }
    int j = i - D_ * NG;
    if (j >= 0 && j < H_ * NG) {             // Wh is [H][NG] row-major
        int k = j / NG, n = j % NG;
        WhT[n * H_ + k] = f2bf(Wh[j]);
    }
}

// ---------------------------------------------------------------------------
// Fused LSTM scan. 16 blocks (one per 16-row batch tile), 512 thr = 16 waves.
// Wave w owns gate columns [w*32, w*32+32). Weights register-resident.
// 2 barriers/step; next x tile prefetched behind the WMMA chain.
// ---------------------------------------------------------------------------
__global__ __launch_bounds__(512) void lstm_scan(
    const float* __restrict__ x,  const float* __restrict__ bh,
    const float* __restrict__ Wd, const float* __restrict__ bd,
    const bf16_t* __restrict__ WiT, const bf16_t* __restrict__ WhT,
    float* __restrict__ out)
{
    __shared__ bf16_t xs[16 * XS_STRIDE];   // x tile, bf16
    __shared__ bf16_t hs[16 * HS_STRIDE];   // h state, bf16 (WMMA A operand)
    __shared__ float  cs[16 * H_];          // c state, f32
    __shared__ float  gates[16 * GS];       // pre-activation gates, f32
    __shared__ float  hf[16 * H_];          // final h_T in f32 for Dense head

    const int tid  = threadIdx.x;
    const int lane = tid & 31;
    const int w    = tid >> 5;              // wave id 0..15
    const int b0   = blockIdx.x * 16;       // batch tile base row

    const int mrow  = lane & 15;
    const int half  = lane >> 4;
    const int nbase = w * 32;

    // ---- one-time: weight B-fragments into registers -----------------------
    v16bf bwi[2][2];                        // [ntile][kb], K=64
    v16bf bwh[2][4];                        // [ntile][kb], K=128
    #pragma unroll
    for (int nt = 0; nt < 2; ++nt) {
        int n = nbase + nt * 16 + mrow;
        #pragma unroll
        for (int kb = 0; kb < 2; ++kb)
            bwi[nt][kb] = *(const v16bf*)(WiT + n * D_ + kb * 32 + half * 16);
        #pragma unroll
        for (int kb = 0; kb < 4; ++kb)
            bwh[nt][kb] = *(const v16bf*)(WhT + n * H_ + kb * 32 + half * 16);
    }

    // ---- per-thread loop-invariant indices / biases ------------------------
    // x staging slot: 2 floats per thread
    const int xm = (tid * 2) >> 6, xd = (tid * 2) & 63;
    const float* xptr = x + ((size_t)(b0 + xm) * T_) * D_ + xd;  // + t*D_ per step
    // elementwise slot: 4 states per thread
    const int em = (tid * 4) >> 7, ej0 = (tid * 4) & 127;
    float bh_i[4], bh_f[4], bh_g[4], bh_o[4];
    #pragma unroll
    for (int jj = 0; jj < 4; ++jj) {
        int j = ej0 + jj;
        bh_i[jj] = bh[j];
        bh_f[jj] = bh[128 + j];
        bh_g[jj] = bh[256 + j];
        bh_o[jj] = bh[384 + j];
    }

    // ---- zero initial state + stage x[:, 0, :] -----------------------------
    for (int i = tid; i < 16 * HS_STRIDE; i += 512) hs[i] = f2bf(0.f);
    for (int i = tid; i < 16 * H_;        i += 512) cs[i] = 0.f;
    {
        const v2f v = __builtin_nontemporal_load((const v2f*)xptr);
        xs[xm * XS_STRIDE + xd]     = f2bf(v.x);
        xs[xm * XS_STRIDE + xd + 1] = f2bf(v.y);
    }
    __syncthreads();

    for (int t = 0; t < T_; ++t) {
        // ---- A fragments from LDS (lanes 0-15: K 0-15; 16-31: 16-31)
        v16bf ax[2], ah[4];
        #pragma unroll
        for (int kb = 0; kb < 2; ++kb) {
            const bf16_t* p = xs + mrow * XS_STRIDE + kb * 32 + half * 16;
            v8bf lo = *(const v8bf*)p;
            v8bf hi = *(const v8bf*)(p + 8);
            #pragma unroll
            for (int j = 0; j < 8; ++j) { ax[kb][j] = lo[j]; ax[kb][j + 8] = hi[j]; }
        }
        #pragma unroll
        for (int kb = 0; kb < 4; ++kb) {
            const bf16_t* p = hs + mrow * HS_STRIDE + kb * 32 + half * 16;
            v8bf lo = *(const v8bf*)p;
            v8bf hi = *(const v8bf*)(p + 8);
            #pragma unroll
            for (int j = 0; j < 8; ++j) { ah[kb][j] = lo[j]; ah[kb][j + 8] = hi[j]; }
        }

        // ---- prefetch next x tile (hidden behind the WMMA chain) -----------
        v2f xnext = {0.f, 0.f};
        if (t + 1 < T_)
            xnext = __builtin_nontemporal_load((const v2f*)(xptr + (t + 1) * D_));

        // ---- gates = x_t @ Wi + h @ Wh  (f32 accumulate, chained C) --------
        v8f acc[2];
        #pragma unroll
        for (int nt = 0; nt < 2; ++nt) {
            v8f a = {};
            #pragma unroll
            for (int kb = 0; kb < 2; ++kb)
                a = __builtin_amdgcn_wmma_f32_16x16x32_bf16(
                        false, ax[kb], false, bwi[nt][kb], (short)0, a, false, false);
            #pragma unroll
            for (int kb = 0; kb < 4; ++kb)
                a = __builtin_amdgcn_wmma_f32_16x16x32_bf16(
                        false, ah[kb], false, bwh[nt][kb], (short)0, a, false, false);
            acc[nt] = a;
        }

        // ---- scatter C/D fragments: M = v + 8*half, N = nbase+nt*16+mrow ---
        #pragma unroll
        for (int nt = 0; nt < 2; ++nt) {
            int n = nbase + nt * 16 + mrow;
            #pragma unroll
            for (int v = 0; v < 8; ++v)
                gates[(v + 8 * half) * GS + n] = acc[nt][v];
        }
        __syncthreads();

        // ---- cell update: branch-free HW transcendentals -------------------
        #pragma unroll
        for (int jj = 0; jj < 4; ++jj) {
            int j = ej0 + jj;
            float si = fast_sigmoid(gates[em * GS + j]       + bh_i[jj]);
            float sf = fast_sigmoid(gates[em * GS + 128 + j] + bh_f[jj]);
            float tg = fast_tanh   (gates[em * GS + 256 + j] + bh_g[jj]);
            float so = fast_sigmoid(gates[em * GS + 384 + j] + bh_o[jj]);
            float c  = sf * cs[em * H_ + j] + si * tg;
            cs[em * H_ + j] = c;
            float h = so * fast_tanh(c);
            hs[em * HS_STRIDE + j] = f2bf(h);
            if (t == T_ - 1) hf[em * H_ + j] = h;
        }

        // ---- commit prefetched x tile for step t+1 -------------------------
        if (t + 1 < T_) {
            xs[xm * XS_STRIDE + xd]     = f2bf(xnext.x);
            xs[xm * XS_STRIDE + xd + 1] = f2bf(xnext.y);
        }
        __syncthreads();
    }

    // ---- Dense(1) + ReLU head ---------------------------------------------
    if (tid < 16) {
        float s = bd[0];
        #pragma unroll 8
        for (int j = 0; j < H_; ++j) s += hf[tid * H_ + j] * Wd[j];
        out[b0 + tid] = s > 0.f ? s : 0.f;
    }
}

extern "C" void kernel_launch(void* const* d_in, const int* in_sizes, int n_in,
                              void* d_out, int out_size, void* d_ws, size_t ws_size,
                              hipStream_t stream) {
    const float* x  = (const float*)d_in[0];
    const float* Wi = (const float*)d_in[1];
    const float* Wh = (const float*)d_in[2];
    const float* bh = (const float*)d_in[3];
    const float* Wd = (const float*)d_in[4];
    const float* bd = (const float*)d_in[5];

    bf16_t* WiT = (bf16_t*)d_ws;            // 32768 bf16 = 64 KB
    bf16_t* WhT = WiT + (size_t)D_ * NG;    // 65536 bf16 = 128 KB

    lstm_prep<<<(D_ * NG + H_ * NG + 255) / 256, 256, 0, stream>>>(Wi, Wh, WiT, WhT);
    lstm_scan<<<B_ / 16, 512, 0, stream>>>(x, bh, Wd, bd, WiT, WhT, (float*)d_out);
}
